// SphereNet_GNN_Interact_41618233099057
// MI455X (gfx1250) — matrix operand order repaired and checked by reading, not compiled, not run on hardware
//
#include <hip/hip_runtime.h>
#include <math.h>

// ---------------------------------------------------------------------------
// MI455X (gfx1250) SphereNet+GINE+FPN forward.
// Dense GEMMs -> v_wmma_f32_16x16x32_bf16 (wave32). Each wave owns a 32x64
// output tile (two 16-row A frags share every B frag -> 8 WMMAs per 4 b128 A
// loads + 64 u16 B loads, halving VMEM issue pressure vs a 16x64 tile).
// Small-K contractions (rbf K=6, sbf/tbf rank-8) -> fused VALU kernels.
// ---------------------------------------------------------------------------

typedef __attribute__((ext_vector_type(16))) __bf16 v16bf;
typedef __attribute__((ext_vector_type(8)))  float  v8f;

__device__ __forceinline__ unsigned short f2bf(float f) {
  unsigned int u = __float_as_uint(f);
  u += 0x7FFFu + ((u >> 16) & 1u);          // round-to-nearest-even
  return (unsigned short)(u >> 16);
}
__device__ __forceinline__ float siluf(float x) { return x / (1.f + __expf(-x)); }

union FragAB { v16bf v; uint4 q[2]; };

// ---------------------------------------------------------------------------
// Tiled bf16 GEMM: C[M,N] = A[M,K] @ B[K,N], f32 accumulate via WMMA.
// block = 128 threads = 4 waves; block tile 128(M) x 64(N); wave tile 32x64.
// N % 64 == 0, K % 32 == 0 (buffers padded by construction); M guarded.
// Epilogue: act(0=none,1=silu,2=relu), then +addsrc, then *mulsrc; writes
// optional f32 master and/or bf16 shadow (for the next GEMM in the chain).
// ---------------------------------------------------------------------------
__global__ __launch_bounds__(128)
void k_gemm_bf16(const unsigned short* __restrict__ A,
                 const unsigned short* __restrict__ Bw,
                 float* __restrict__ Cf, unsigned short* __restrict__ Cb,
                 const float* __restrict__ addsrc, const float* __restrict__ mulsrc,
                 int M, int N, int K, int act)
{
  const int lane = threadIdx.x & 31;
  const int wave = threadIdx.x >> 5;
  const int rowBase = blockIdx.y * 128 + wave * 32;
  const int colBase = blockIdx.x * 64;

  v8f acc[2][4];
  #pragma unroll
  for (int h = 0; h < 2; ++h)
    #pragma unroll
    for (int nt = 0; nt < 4; ++nt)
      #pragma unroll
      for (int i = 0; i < 8; ++i) acc[h][nt][i] = 0.f;

  // ISA 7.12.2: 16-bit A 16x32 -- lane m: K{0..7,16..23}; lane m+16: K{8..15,24..31}
  const int koff = (lane < 16) ? 0 : 8;
  int ar0 = rowBase + (lane & 15);
  int ar1 = ar0 + 16;
  ar0 = (ar0 < M - 1) ? ar0 : (M - 1);   // clamp (rows >= M are discarded below)
  ar1 = (ar1 < M - 1) ? ar1 : (M - 1);

  for (int k0 = 0; k0 < K; k0 += 32) {
    FragAB a0, a1;
    const uint4* ap0 = reinterpret_cast<const uint4*>(A + (size_t)ar0 * K + k0 + koff);
    a0.q[0] = ap0[0];            // K = k0+koff   .. +7
    a0.q[1] = ap0[2];            // K = k0+koff+16.. +23
    const uint4* ap1 = reinterpret_cast<const uint4*>(A + (size_t)ar1 * K + k0 + koff);
    a1.q[0] = ap1[0];
    a1.q[1] = ap1[2];
    // B: lane holds column N = colBase+nt*16+(lane&15); K in same order as A
    const unsigned short* bp = Bw + (size_t)(k0 + koff) * N + colBase + (lane & 15);
    #pragma unroll
    for (int nt = 0; nt < 4; ++nt) {
      FragAB b;
      #pragma unroll
      for (int s = 0; s < 8; ++s) {
        b.v[s]     = __builtin_bit_cast(__bf16, bp[(size_t)s * N + nt * 16]);
        b.v[s + 8] = __builtin_bit_cast(__bf16, bp[(size_t)(16 + s) * N + nt * 16]);
      }
      acc[0][nt] = __builtin_amdgcn_wmma_f32_16x16x32_bf16(
          false, a0.v, false, b.v, (short)0, acc[0][nt], false, false);
      acc[1][nt] = __builtin_amdgcn_wmma_f32_16x16x32_bf16(
          false, a1.v, false, b.v, (short)0, acc[1][nt], false, false);
    }
  }

  // C/D layout: lane<16 -> N=lane, M=v; lane>=16 -> N=lane-16, M=8+v
  const int nCol0 = colBase + (lane & 15);
  #pragma unroll
  for (int h = 0; h < 2; ++h) {
    const int mBase = rowBase + h * 16 + ((lane >> 4) << 3);
    #pragma unroll
    for (int nt = 0; nt < 4; ++nt) {
      const int col = nCol0 + nt * 16;
      #pragma unroll
      for (int v = 0; v < 8; ++v) {
        const int row = mBase + v;
        if (row >= M) continue;
        const size_t idx = (size_t)row * N + col;
        float x = acc[h][nt][v];
        if (act == 1) x = siluf(x);
        else if (act == 2) x = fmaxf(x, 0.f);
        if (addsrc) x += addsrc[idx];
        if (mulsrc) x *= mulsrc[idx];
        if (Cf) Cf[idx] = x;
        if (Cb) Cb[idx] = f2bf(x);
      }
    }
  }
}

// f32 -> bf16 with row/col zero-padding (also used for weight conversion)
__global__ void k_cvt_bf16_pad(const float* __restrict__ src, unsigned short* __restrict__ dst,
                               int R, int C, int Rpad, int Cpad)
{
  long long i = (long long)blockIdx.x * blockDim.x + threadIdx.x;
  long long total = (long long)Rpad * Cpad;
  if (i >= total) return;
  int r = (int)(i / Cpad), c = (int)(i % Cpad);
  dst[i] = (r < R && c < C) ? f2bf(src[(size_t)r * C + c]) : (unsigned short)0;
}

// tiny f32 GEMM (used once: Wc[6,128] = Wi_rbf0 @ Wi_cat[256:384])
__global__ void k_small_mm(const float* __restrict__ A, const float* __restrict__ B,
                           float* __restrict__ C, int M, int K, int N)
{
  int i = blockIdx.x * blockDim.x + threadIdx.x;
  if (i >= M * N) return;
  int r = i / N, c = i % N;
  float s = 0.f;
  for (int k = 0; k < K; ++k) s += A[r * K + k] * B[(size_t)k * N + c];
  C[i] = s;
}

// per-edge geometry: vec, dist, Bessel rbf with polynomial envelope
__global__ void k_geom(const float* __restrict__ pos, const int* __restrict__ src,
                       const int* __restrict__ dst, float* __restrict__ vec,
                       float* __restrict__ rbf, int E)
{
  int e = blockIdx.x * blockDim.x + threadIdx.x;
  if (e >= E) return;
  int s = src[e], d = dst[e];
  float vx = pos[d * 3 + 0] - pos[s * 3 + 0];
  float vy = pos[d * 3 + 1] - pos[s * 3 + 1];
  float vz = pos[d * 3 + 2] - pos[s * 3 + 2];
  float dist = fmaxf(sqrtf(vx * vx + vy * vy + vz * vz), 0.1f);
  vec[e * 3 + 0] = vx; vec[e * 3 + 1] = vy; vec[e * 3 + 2] = vz;
  float x = dist * (1.f / 5.0f);
  float env = 0.f;
  if (x < 1.f) {
    float x4 = x * x * x * x;
    env = 1.f / x - 21.f * x4 + 35.f * x4 * x - 15.f * x4 * x * x;  // P=5 envelope
  }
  const float pi = 3.14159265358979f;
  #pragma unroll
  for (int n = 0; n < 6; ++n) rbf[e * 6 + n] = env * __sinf((float)(n + 1) * pi * x);
}

// per-triplet angles -> cos(l*angle), cos(l*torsion)
__global__ void k_triplet(const float* __restrict__ vec, const int* __restrict__ idx_ji,
                          const int* __restrict__ idx_kj, float* __restrict__ ca,
                          float* __restrict__ ct, int T)
{
  int t = blockIdx.x * blockDim.x + threadIdx.x;
  if (t >= T) return;
  int ji = idx_ji[t], kj = idx_kj[t];
  float ax = vec[ji * 3], ay = vec[ji * 3 + 1], az = vec[ji * 3 + 2];
  float bx = vec[kj * 3], by = vec[kj * 3 + 1], bz = vec[kj * 3 + 2];
  float cx = ay * bz - az * by, cy = az * bx - ax * bz, cz = ax * by - ay * bx;
  float crn = sqrtf(cx * cx + cy * cy + cz * cz) + 1e-9f;
  float dot = ax * bx + ay * by + az * bz;
  float angle = atan2f(crn, dot);
  float tors  = atan2f(cy, cx + 1e-9f);
  #pragma unroll
  for (int s = 0; s < 7; ++s) {
    ca[t * 7 + s] = __cosf((float)s * angle);
    ct[t * 7 + s] = __cosf((float)s * tors);
  }
}

// e1 = silu( Tdst[z[dst]] + Tsrc[z[src]] + rbf @ Wc ), fused (replaces h0@Wi_cat)
__global__ __launch_bounds__(128)
void k_e1_init(const float* __restrict__ rbf, const int* __restrict__ z,
               const int* __restrict__ src, const int* __restrict__ dst,
               const float* __restrict__ Tdst, const float* __restrict__ Tsrc,
               const float* __restrict__ Wc, float* __restrict__ e1f,
               unsigned short* __restrict__ e1b)
{
  int e = blockIdx.x, j = threadIdx.x;
  __shared__ float r6[6];
  __shared__ int zd, zs;
  if (j < 6) r6[j] = rbf[e * 6 + j];
  if (j == 6) zd = z[dst[e]];
  if (j == 7) zs = z[src[e]];
  __syncthreads();
  float g = Tdst[zd * 128 + j] + Tsrc[zs * 128 + j];
  #pragma unroll
  for (int r = 0; r < 6; ++r) g += r6[r] * Wc[r * 128 + j];
  g = siluf(g);
  size_t idx = (size_t)e * 128 + j;
  e1f[idx] = g; e1b[idx] = f2bf(g);
}

// out = (rbf @ W[6,128]) * in      (Wi_rbf1 / We_rbf_e2 gates)
__global__ __launch_bounds__(128)
void k_gate6_mul(const float* __restrict__ rbf, const float* __restrict__ W,
                 const float* __restrict__ in, float* __restrict__ outf,
                 unsigned short* __restrict__ outb)
{
  int e = blockIdx.x, j = threadIdx.x;
  __shared__ float r6[6];
  if (j < 6) r6[j] = rbf[e * 6 + j];
  __syncthreads();
  float g = 0.f;
  #pragma unroll
  for (int r = 0; r < 6; ++r) g += r6[r] * W[r * 128 + j];
  size_t idx = (size_t)e * 128 + j;
  float v = g * in[idx];
  if (outf) outf[idx] = v;
  if (outb) outb[idx] = f2bf(v);
}

// out = ((rbf @ W1[6,8]) @ W2[8,128]) * in   (rank-8 rbf gate on x_kj)
__global__ __launch_bounds__(128)
void k_gate_rank8_mul(const float* __restrict__ rbf, const float* __restrict__ W1,
                      const float* __restrict__ W2, const float* __restrict__ in,
                      float* __restrict__ outf, unsigned short* __restrict__ outb)
{
  int e = blockIdx.x, j = threadIdx.x;
  __shared__ float r6[6], h8[8];
  if (j < 6) r6[j] = rbf[e * 6 + j];
  __syncthreads();
  if (j < 8) {
    float s = 0.f;
    #pragma unroll
    for (int r = 0; r < 6; ++r) s += r6[r] * W1[r * 8 + j];
    h8[j] = s;
  }
  __syncthreads();
  float g = 0.f;
  #pragma unroll
  for (int i = 0; i < 8; ++i) g += h8[i] * W2[i * 128 + j];
  size_t idx = (size_t)e * 128 + j;
  float v = g * in[idx];
  outf[idx] = v;
  if (outb) outb[idx] = f2bf(v);
}

// per-triplet: out[t,:] = xd[idx_kj[t],:] * (sbf@Ws1@Ws2) * (tbf@Wt1@Wt2)
// rank-8 factorization kept (42->8->64, 294->8->64): ~3K MACs/triplet.
__global__ __launch_bounds__(64)
void k_triplet_mix(const float* __restrict__ xd, const float* __restrict__ ca,
                   const float* __restrict__ ct, const float* __restrict__ rbf,
                   const int* __restrict__ idx_kj,
                   const float* __restrict__ Ws1, const float* __restrict__ Ws2,
                   const float* __restrict__ Wt1, const float* __restrict__ Wt2,
                   float* __restrict__ out)
{
  int t = blockIdx.x, j = threadIdx.x;
  __shared__ float sca[7], sct[7], srad[6], sw[42], h8s[8], h8t[8];
  __shared__ int skj;
  if (j == 0) skj = idx_kj[t];
  if (j < 7) sca[j] = ca[t * 7 + j];
  if (j >= 16 && j < 23) sct[j - 16] = ct[t * 7 + (j - 16)];
  __syncthreads();
  if (j < 6) srad[j] = rbf[(size_t)skj * 6 + j];
  __syncthreads();
  if (j < 42) sw[j] = sct[j / 6] * srad[j % 6];
  __syncthreads();
  if (j < 8) {
    float s = 0.f;
    for (int q = 0; q < 42; ++q) s += sca[q / 6] * srad[q % 6] * Ws1[q * 8 + j];
    h8s[j] = s;
  } else if (j < 16) {
    int i = j - 8;
    float s = 0.f;
    for (int s1 = 0; s1 < 7; ++s1) {
      float c = sca[s1];
      const float* w = Wt1 + (size_t)(s1 * 42) * 8;
      for (int q = 0; q < 42; ++q) s += c * sw[q] * w[q * 8 + i];
    }
    h8t[i] = s;
  }
  __syncthreads();
  float sv = 0.f, tv = 0.f;
  #pragma unroll
  for (int i = 0; i < 8; ++i) {
    sv += h8s[i] * Ws2[i * 64 + j];
    tv += h8t[i] * Wt2[i * 64 + j];
  }
  out[(size_t)t * 64 + j] = xd[(size_t)skj * 64 + j] * sv * tv;
}

// seg(x_kj, idx_ji): idx_ji = repeat(arange(E),2)  ->  acc[e] = tm[2e] + tm[2e+1]
__global__ void k_pair_sum(const float* __restrict__ tm, float* __restrict__ outf,
                           unsigned short* __restrict__ outb, int E)
{
  long long i = (long long)blockIdx.x * blockDim.x + threadIdx.x;
  if (i >= (long long)E * 64) return;
  long long e = i / 64; int j = (int)(i % 64);
  float v = tm[(size_t)(2 * e) * 64 + j] + tm[(size_t)(2 * e + 1) * 64 + j];
  outf[i] = v; outb[i] = f2bf(v);
}

__global__ void k_add2(const float* __restrict__ a, const float* __restrict__ b,
                       float* __restrict__ outf, unsigned short* __restrict__ outb,
                       long long n)
{
  long long i = (long long)blockIdx.x * blockDim.x + threadIdx.x;
  if (i >= n) return;
  float v = a[i] + b[i];
  outf[i] = v;
  if (outb) outb[i] = f2bf(v);
}

// seg(e2, sph_dst): sph_dst = repeat(arange(N),16) -> fixed fan-in 16 reduction
__global__ void k_deg16_sum(const float* __restrict__ e2, unsigned short* __restrict__ outb,
                            int N, int Npad)
{
  long long i = (long long)blockIdx.x * blockDim.x + threadIdx.x;
  if (i >= (long long)Npad * 128) return;
  int n = (int)(i / 128), j = (int)(i % 128);
  float s = 0.f;
  if (n < N)
    for (int k = 0; k < 16; ++k) s += e2[(size_t)(n * 16 + k) * 128 + j];
  outb[i] = f2bf(s);
}

// batch pooling: batch_seg is sorted -> binary search the contiguous range
__global__ __launch_bounds__(128)
void k_batch_pool(const float* __restrict__ v, const int* __restrict__ bseg,
                  float* __restrict__ out, int N, int accumulate)
{
  int b = blockIdx.x, j = threadIdx.x;
  int lo = 0, hi = N;
  while (lo < hi) { int m = (lo + hi) >> 1; if (bseg[m] < b) lo = m + 1; else hi = m; }
  int start = lo;
  lo = 0; hi = N;
  while (lo < hi) { int m = (lo + hi) >> 1; if (bseg[m] < b + 1) lo = m + 1; else hi = m; }
  int end = lo;
  float s = 0.f;
  for (int n = start; n < end; ++n) s += v[(size_t)n * 128 + j];
  if (accumulate) out[b * 128 + j] += s; else out[b * 128 + j] = s;
}

// GINE aggregation: dst2 = repeat(arange(N),4) -> fixed fan-in 4
__global__ void k_gine_agg(const float* __restrict__ hid, const float* __restrict__ eaw,
                           const int* __restrict__ src2, const float* __restrict__ eps_g,
                           int layer, unsigned short* __restrict__ preb, int N, int Npad)
{
  long long i = (long long)blockIdx.x * blockDim.x + threadIdx.x;
  if (i >= (long long)Npad * 128) return;
  int n = (int)(i / 128), j = (int)(i % 128);
  float v = 0.f;
  if (n < N) {
    v = (1.f + eps_g[layer]) * hid[(size_t)n * 128 + j];
    #pragma unroll
    for (int k = 0; k < 4; ++k) {
      int e = n * 4 + k;
      float m = hid[(size_t)src2[e] * 128 + j] + eaw[(size_t)e * 128 + j];
      v += fmaxf(m, 0.f);
    }
  }
  preb[i] = f2bf(v);
}

__global__ void k_concat3(const float* __restrict__ a, const float* __restrict__ b,
                          const float* __restrict__ c, unsigned short* __restrict__ cat)
{
  int i = blockIdx.x * blockDim.x + threadIdx.x;
  if (i >= 64 * 384) return;
  int row = i / 384, col = i % 384;
  float v = (col < 128) ? a[row * 128 + col]
          : (col < 256) ? b[row * 128 + col - 128]
                        : c[row * 128 + col - 256];
  cat[i] = f2bf(v);
}

__global__ __launch_bounds__(64)
void k_out_head(const float* __restrict__ ch, const float* __restrict__ Wout,
                float* __restrict__ out)
{
  int b = threadIdx.x;
  float s = 0.f;
  for (int j = 0; j < 128; ++j) s += ch[b * 128 + j] * Wout[j];
  out[b] = s;
}

__global__ void k_copy_f32(const float* __restrict__ src, float* __restrict__ dst, long long n)
{
  long long i = (long long)blockIdx.x * blockDim.x + threadIdx.x;
  if (i < n) dst[i] = src[i];
}

// ---------------------------------------------------------------------------
// Host orchestration
// ---------------------------------------------------------------------------
namespace {
struct Arena {
  char* base; size_t off; size_t cap;
  void* alloc(size_t bytes) {
    size_t a = (off + 255) & ~size_t(255);
    off = a + bytes;
    return base + a;
  }
};
}

extern "C" void kernel_launch(void* const* d_in, const int* in_sizes, int n_in,
                              void* d_out, int out_size, void* d_ws, size_t ws_size,
                              hipStream_t stream)
{
  (void)in_sizes; (void)n_in; (void)out_size;
  const int N = 10000, Npad = 10048, E = 160000, T = 320000, E2 = 40000, H = 128;

  const float* pos        = (const float*)d_in[0];
  const int*   z          = (const int*)d_in[1];
  const int*   batch_seg  = (const int*)d_in[2];
  const int*   sph_src    = (const int*)d_in[3];
  const int*   sph_dst    = (const int*)d_in[4];
  const int*   idx_kj     = (const int*)d_in[5];
  const int*   idx_ji     = (const int*)d_in[6];
  const float* x_feat     = (const float*)d_in[7];
  const int*   src2       = (const int*)d_in[8];
  const float* edge_attr  = (const float*)d_in[10];
  const float* fingerpr   = (const float*)d_in[11];
  const float* atom_emb   = (const float*)d_in[12];
  const float* Wi_rbf0    = (const float*)d_in[13];
  const float* Wi_cat     = (const float*)d_in[14];
  const float* Wi_rbf1    = (const float*)d_in[15];
  const float* Wv_up      = (const float*)d_in[16];
  const float* Wv_hid     = (const float*)d_in[17];
  const float* Wv_out     = (const float*)d_in[18];
  const float* We_ji      = (const float*)d_in[19];
  const float* We_kj      = (const float*)d_in[20];
  const float* We_rbf1    = (const float*)d_in[21];
  const float* We_rbf2    = (const float*)d_in[22];
  const float* We_sbf1    = (const float*)d_in[23];
  const float* We_sbf2    = (const float*)d_in[24];
  const float* We_t1      = (const float*)d_in[25];
  const float* We_t2      = (const float*)d_in[26];
  const float* We_down    = (const float*)d_in[27];
  const float* We_up      = (const float*)d_in[28];
  const float* We_res     = (const float*)d_in[29];
  const float* We_rbf_e2  = (const float*)d_in[30];
  const float* W_n2h      = (const float*)d_in[31];
  const float* W_edge     = (const float*)d_in[32];
  const float* eps_g      = (const float*)d_in[33];
  const float* W_nn       = (const float*)d_in[34];
  const float* Wf1        = (const float*)d_in[35];
  const float* Wf2        = (const float*)d_in[36];
  const float* W_hid      = (const float*)d_in[37];
  const float* W_out      = (const float*)d_in[38];
  float* out = (float*)d_out;

  Arena ar{(char*)d_ws, 0, ws_size};
  auto F  = [&](size_t n) { return (float*)ar.alloc(n * sizeof(float)); };
  auto Hb = [&](size_t n) { return (unsigned short*)ar.alloc(n * sizeof(unsigned short)); };

  // bf16 weight shadows (K padded to 32 where needed)
  unsigned short* wb_Wi_cat = Hb(384 * 128);
  unsigned short* wb_emb    = Hb(128 * 128);
  unsigned short* wb_Wv_up  = Hb(5 * 128 * 256);
  unsigned short* wb_Wv_hid = Hb(15 * 256 * 256);
  unsigned short* wb_Wv_out = Hb(5 * 256 * 128);
  unsigned short* wb_We_ji  = Hb(4 * 128 * 128);
  unsigned short* wb_We_kj  = Hb(4 * 128 * 128);
  unsigned short* wb_We_dn  = Hb(4 * 128 * 64);
  unsigned short* wb_We_up  = Hb(4 * 64 * 128);
  unsigned short* wb_We_res = Hb(4 * 7 * 128 * 128);
  unsigned short* wb_W_n2h  = Hb(64 * 128);
  unsigned short* wb_W_edge = Hb(4 * 32 * 128);
  unsigned short* wb_W_nn   = Hb(128 * 128);
  unsigned short* wb_Wf1    = Hb(2048 * 128);
  unsigned short* wb_Wf2    = Hb(128 * 128);
  unsigned short* wb_W_hid  = Hb(384 * 128);

  // activations / geometry
  float* VEC = F((size_t)E * 3);
  float* RBF = F((size_t)E * 6);
  float* CA  = F((size_t)T * 7);
  float* CT  = F((size_t)T * 7);
  float* Ae  = F((size_t)E * H);  unsigned short* Ab  = Hb((size_t)E * H);   // e1
  float* C1  = F((size_t)E * H);                                             // x_ji
  float* C2  = F((size_t)E * H);  unsigned short* C2b = Hb((size_t)E * H);
  float* C3  = F((size_t)E * H);  unsigned short* C3b = Hb((size_t)E * H);   // h / e2
  float* D64 = F((size_t)E * 64);
  float* TM  = F((size_t)T * 64);
  float* S64 = F((size_t)E * 64); unsigned short* S64b = Hb((size_t)E * 64);
  unsigned short* VSEGB = Hb((size_t)Npad * H);
  unsigned short* V1b   = Hb((size_t)Npad * 256);
  unsigned short* V2b   = Hb((size_t)Npad * 256);
  float* VOUT = F((size_t)Npad * H);
  float* U    = F(64 * H);
  float* GP   = F(64 * H);
  float* FCN  = F(64 * H);
  float* CH   = F(64 * H);
  float* TDST = F(128 * 128);
  float* TSRC = F(128 * 128);
  float* WC   = F(6 * 128);
  unsigned short* XFB  = Hb((size_t)Npad * 64);
  float* HID  = F((size_t)Npad * H);
  unsigned short* PREb = Hb((size_t)Npad * H);
  float* EAW  = F((size_t)E2 * H);
  float* SPH  = F((size_t)Npad * H); unsigned short* SPHb = Hb((size_t)Npad * H);
  unsigned short* EAB  = Hb((size_t)E2 * 32);
  unsigned short* FPB  = Hb(64 * 2048);
  unsigned short* F1B  = Hb(64 * 128);
  unsigned short* CATB = Hb(64 * 384);

  auto cvt = [&](const float* s, unsigned short* d, int R, int C, int Rp, int Cp) {
    long long tot = (long long)Rp * Cp;
    k_cvt_bf16_pad<<<dim3((unsigned)((tot + 255) / 256)), 256, 0, stream>>>(s, d, R, C, Rp, Cp);
  };
  auto gemm = [&](const unsigned short* A, const unsigned short* Bw, float* Cf,
                  unsigned short* Cb, const float* add, const float* mul,
                  int M, int Nc, int K, int act) {
    dim3 g(Nc / 64, (M + 127) / 128);
    k_gemm_bf16<<<g, 128, 0, stream>>>(A, Bw, Cf, Cb, add, mul, M, Nc, K, act);
  };
  auto g1d = [&](long long n) { return dim3((unsigned)((n + 255) / 256)); };

  // ---- weight conversions ----
  cvt(Wi_cat, wb_Wi_cat, 384, 128, 384, 128);
  cvt(atom_emb, wb_emb, 95, 128, 128, 128);
  cvt(Wv_up, wb_Wv_up, 640, 256, 640, 256);
  cvt(Wv_hid, wb_Wv_hid, 3840, 256, 3840, 256);
  cvt(Wv_out, wb_Wv_out, 1280, 128, 1280, 128);
  cvt(We_ji, wb_We_ji, 512, 128, 512, 128);
  cvt(We_kj, wb_We_kj, 512, 128, 512, 128);
  cvt(We_down, wb_We_dn, 512, 64, 512, 64);
  cvt(We_up, wb_We_up, 256, 128, 256, 128);
  cvt(We_res, wb_We_res, 3584, 128, 3584, 128);
  cvt(W_n2h, wb_W_n2h, 39, 128, 64, 128);
  for (int l = 0; l < 4; ++l)
    cvt(W_edge + (size_t)l * 10 * 128, wb_W_edge + (size_t)l * 32 * 128, 10, 128, 32, 128);
  cvt(W_nn, wb_W_nn, 128, 128, 128, 128);
  cvt(Wf1, wb_Wf1, 2048, 128, 2048, 128);
  cvt(Wf2, wb_Wf2, 128, 128, 128, 128);
  cvt(W_hid, wb_W_hid, 384, 128, 384, 128);
  cvt(x_feat, XFB, N, 39, Npad, 64);
  cvt(edge_attr, EAB, E2, 10, E2, 32);
  cvt(fingerpr, FPB, 64, 2048, 64, 2048);

  // ---- atom-type tables (WMMA) + fused rbf projection weight ----
  gemm(wb_emb, wb_Wi_cat, TDST, nullptr, nullptr, nullptr, 128, 128, 128, 0);
  gemm(wb_emb, wb_Wi_cat + 128 * 128, TSRC, nullptr, nullptr, nullptr, 128, 128, 128, 0);
  k_small_mm<<<g1d(6 * 128), 256, 0, stream>>>(Wi_rbf0, Wi_cat + 256 * 128, WC, 6, 128, 128);

  // ---- geometry ----
  k_geom<<<g1d(E), 256, 0, stream>>>(pos, sph_src, sph_dst, VEC, RBF, E);
  k_triplet<<<g1d(T), 256, 0, stream>>>(VEC, idx_ji, idx_kj, CA, CT, T);

  // ---- init embedding: e1, e2 ----
  k_e1_init<<<E, 128, 0, stream>>>(RBF, z, sph_src, sph_dst, TDST, TSRC, WC, Ae, Ab);
  k_gate6_mul<<<E, 128, 0, stream>>>(RBF, Wi_rbf1, Ae, C3, nullptr);

  auto run_update_v = [&](int l) {
    k_deg16_sum<<<g1d((long long)Npad * H), 256, 0, stream>>>(C3, VSEGB, N, Npad);
    gemm(VSEGB, wb_Wv_up + (size_t)l * H * 256, nullptr, V1b, nullptr, nullptr, Npad, 256, H, 0);
    unsigned short* pin = V1b; unsigned short* pout = V2b;
    for (int k = 0; k < 3; ++k) {
      gemm(pin, wb_Wv_hid + (size_t)(l * 3 + k) * 256 * 256, nullptr, pout,
           nullptr, nullptr, Npad, 256, 256, 1);
      unsigned short* t = pin; pin = pout; pout = t;
    }
    gemm(pin, wb_Wv_out + (size_t)l * 256 * H, VOUT, nullptr, nullptr, nullptr, Npad, H, 256, 0);
  };

  auto run_update_e = [&](int l) {
    gemm(Ab, wb_We_ji + (size_t)l * H * H, C1, nullptr, nullptr, nullptr, E, H, H, 1);
    gemm(Ab, wb_We_kj + (size_t)l * H * H, C2, nullptr, nullptr, nullptr, E, H, H, 1);
    k_gate_rank8_mul<<<E, 128, 0, stream>>>(RBF, We_rbf1 + l * 48, We_rbf2 + l * 8 * 128,
                                            C2, C2, C2b);
    gemm(C2b, wb_We_dn + (size_t)l * H * 64, D64, nullptr, nullptr, nullptr, E, 64, H, 1);
    k_triplet_mix<<<T, 64, 0, stream>>>(D64, CA, CT, RBF, idx_kj,
                                        We_sbf1 + l * 42 * 8, We_sbf2 + l * 8 * 64,
                                        We_t1 + l * 294 * 8, We_t2 + l * 8 * 64, TM);
    k_pair_sum<<<g1d((long long)E * 64), 256, 0, stream>>>(TM, S64, S64b, E);
    gemm(S64b, wb_We_up + (size_t)l * 64 * H, C2, nullptr, nullptr, nullptr, E, H, 64, 1);
    k_add2<<<g1d((long long)E * H), 256, 0, stream>>>(C1, C2, C3, C3b, (long long)E * H);
    const size_t rs = (size_t)H * H;
    unsigned short* Wr = wb_We_res + (size_t)l * 7 * rs;
    gemm(C3b, Wr + 0 * rs, nullptr, C2b, nullptr, nullptr, E, H, H, 1);
    gemm(C2b, Wr + 1 * rs, C3, C3b, C3, nullptr, E, H, H, 1);      // h += silu(silu(h W0) W1)
    gemm(C3b, Wr + 2 * rs, Ae, Ab, Ae, nullptr, E, H, H, 1);       // skip: silu(h W2) + x1
    gemm(Ab,  Wr + 3 * rs, nullptr, C2b, nullptr, nullptr, E, H, H, 1);
    gemm(C2b, Wr + 4 * rs, Ae, Ab, Ae, nullptr, E, H, H, 1);
    gemm(Ab,  Wr + 5 * rs, nullptr, C2b, nullptr, nullptr, E, H, H, 1);
    gemm(C2b, Wr + 6 * rs, Ae, Ab, Ae, nullptr, E, H, H, 1);       // new e1 in Ae/Ab
    k_gate6_mul<<<E, 128, 0, stream>>>(RBF, We_rbf_e2 + (size_t)l * 6 * 128, Ae, C3, nullptr);
  };

  auto run_gine = [&](const float* hidF, float* outF, int l) {
    gemm(EAB, wb_W_edge + (size_t)l * 32 * H, EAW, nullptr, nullptr, nullptr, E2, H, 32, 0);
    k_gine_agg<<<g1d((long long)Npad * H), 256, 0, stream>>>(hidF, EAW, src2, eps_g, l,
                                                             PREb, N, Npad);
    gemm(PREb, wb_W_nn, outF, nullptr, nullptr, nullptr, Npad, H, H, 2);
  };

  // ---- SphereNet main loop ----
  run_update_v(0);
  k_batch_pool<<<64, 128, 0, stream>>>(VOUT, batch_seg, U, N, 0);
  for (int l = 0; l < 4; ++l) {
    run_update_e(l);
    run_update_v(l + 1);
    k_batch_pool<<<64, 128, 0, stream>>>(VOUT, batch_seg, U, N, 1);
  }

  // ---- GINE chain (3 layers) ----
  gemm(XFB, wb_W_n2h, HID, nullptr, nullptr, nullptr, Npad, H, 64, 0);
  for (int i = 0; i < 3; ++i) run_gine(HID, HID, i);

  // ---- FPN ----
  gemm(FPB, wb_Wf1, nullptr, F1B, nullptr, nullptr, 64, 128, 2048, 2);
  gemm(F1B, wb_Wf2, FCN, nullptr, nullptr, nullptr, 64, 128, 128, 0);

  // ---- sphere_out = v + gnn_out; gnn_out := sphere_out ----
  k_add2<<<g1d((long long)Npad * H), 256, 0, stream>>>(VOUT, HID, SPH, SPHb,
                                                       (long long)Npad * H);

  // ---- extra application of last sphere layer ----
  run_update_e(3);                 // e2f -> C3 (overwrites Ae/Ab, no longer needed)
  run_update_v(4);                 // sphere_v -> VOUT
  k_batch_pool<<<64, 128, 0, stream>>>(VOUT, batch_seg, U, N, 1);   // sphere_pool in U

  // ---- final GINE layer on sphere_out, pool ----
  run_gine(SPH, HID, 3);
  k_batch_pool<<<64, 128, 0, stream>>>(HID, batch_seg, GP, N, 0);   // gnn_pool

  // ---- head: cat -> relu(cat@W_hid) -> @W_out ----
  k_concat3<<<g1d(64 * 384), 256, 0, stream>>>(GP, FCN, U, CATB);
  gemm(CATB, wb_W_hid, CH, nullptr, nullptr, nullptr, 64, 128, 384, 2);
  k_out_head<<<1, 64, 0, stream>>>(CH, W_out, out);
  k_copy_f32<<<g1d(64 * 128), 256, 0, stream>>>(GP, out + 64, 64 * 128);
  k_copy_f32<<<g1d(64 * 128), 256, 0, stream>>>(U, out + 64 + 64 * 128, 64 * 128);
}
